// SequenceGenerator_2010044694902
// MI455X (gfx1250) — matrix operand order repaired and checked by reading, not compiled, not run on hardware
//
#include <hip/hip_runtime.h>
#include <hip/hip_bf16.h>
#include <math.h>

// ---------- types for WMMA ----------
typedef __attribute__((ext_vector_type(16))) __bf16     v16bf;
typedef __attribute__((ext_vector_type(8)))  float      v8f;
typedef __attribute__((ext_vector_type(4)))  unsigned int v4u;

union ABFrag { v4u u[2]; v16bf v; };

__device__ __forceinline__ unsigned short f2bf(float f) {
    unsigned int u = __float_as_uint(f);
    u += 0x7FFFu + ((u >> 16) & 1u);   // round-to-nearest-even
    return (unsigned short)(u >> 16);
}

__device__ __forceinline__ float hsig(float t) {
    float v = 0.2f * t + 0.5f;
    v = v < 0.f ? 0.f : v;
    return v > 1.f ? 1.f : v;
}

// Problem constants (from reference)
#define HH   48
#define WW   48
#define CC   64
#define HID  64
#define HP   50          // padded
#define WP   50
#define HORIZON 108
#define KG   1152        // gate GEMM K = 9 taps * (64 x + 64 h)
#define NG   256         // gate GEMM N = 4*HID
#define KY   576         // out GEMM K = 9 taps * 64
#define NY   64

// ---------------------------------------------------------------------------
// Weight repack: gate weights [Wx | Wh] -> fragment-major bf16.
// Fragment (kt, nt): 32 lanes x 16 bf16, contiguous.
// B fragment striping (ISA 7.12.2): lane<16 -> K 0..7,16..23, lane>=16 -> K 8..15,24..31,
// column n = nt*16 + (lane&15); per lane, slot s -> K = kt*32 + 8*(lane>=16) + (s&7) + 2*(s&8).
// ---------------------------------------------------------------------------
__global__ void pack_gate_w(const float* __restrict__ Wx,
                            const float* __restrict__ Wh,
                            unsigned short* __restrict__ Wg) {
    int idx = blockIdx.x * 256 + threadIdx.x;
    if (idx >= KG * NG) return;
    int f    = idx >> 9;        // 512 elements per fragment
    int lane = (idx >> 4) & 31;
    int s    = idx & 15;
    int kt   = f >> 4;          // NT = 16
    int nt   = f & 15;
    int n    = nt * 16 + (lane & 15);
    int k    = kt * 32 + ((lane & 16) >> 1) + (s & 7) + ((s & 8) << 1);
    int tap  = k >> 7;
    int ch   = k & 127;
    float v  = (ch < 64) ? Wx[(tap * 64 + ch) * NG + n]
                         : Wh[(tap * 64 + (ch - 64)) * NG + n];
    Wg[idx] = f2bf(v);
}

__global__ void pack_out_w(const float* __restrict__ Cw,
                           unsigned short* __restrict__ Wy) {
    int idx = blockIdx.x * 256 + threadIdx.x;
    if (idx >= KY * NY) return;
    int f    = idx >> 9;
    int lane = (idx >> 4) & 31;
    int s    = idx & 15;
    int kt   = f >> 2;          // NT = 4
    int nt   = f & 3;
    int n    = nt * 16 + (lane & 15);
    int k    = kt * 32 + ((lane & 16) >> 1) + (s & 7) + ((s & 8) << 1);
    int tap  = k >> 6;
    int ch   = k & 63;
    Wy[idx] = f2bf(Cw[(tap * 64 + ch) * NY + n]);
}

// ---------------------------------------------------------------------------
// State init: halo-padded bf16 x / h buffers (zero halo), f32 c buffer.
// ---------------------------------------------------------------------------
__global__ void init_state(const float* __restrict__ x0,
                           const float* __restrict__ h0,
                           const float* __restrict__ c0,
                           unsigned short* __restrict__ xp,
                           unsigned short* __restrict__ hp0,
                           unsigned short* __restrict__ hp1,
                           float* __restrict__ cb) {
    int i = blockIdx.x * 256 + threadIdx.x;
    if (i < HP * WP * CC) {
        int ch  = i & 63;
        int pc  = i >> 6;
        int col = pc % WP;
        int row = pc / WP;
        bool in = (row >= 1 && row <= HH && col >= 1 && col <= WW);
        int si  = in ? (((row - 1) * WW + (col - 1)) * CC + ch) : 0;
        xp[i]  = in ? f2bf(x0[si]) : (unsigned short)0;
        hp0[i] = in ? f2bf(h0[si]) : (unsigned short)0;
        hp1[i] = 0;
    }
    if (i < HH * WW * HID) cb[i] = c0[i];
}

// ---------------------------------------------------------------------------
// Gate kernel: z = conv(x,Wx)+conv(h,Wh)+bias via implicit GEMM (WMMA bf16),
// then LSTM pointwise. Block = 16-pixel row segment; 8 waves each own 32 of
// the 256 gate channels. The 3x18x128-channel halo tile ([x|h]) is staged
// once per block in LDS so A fragments are served by ds_load_b128 instead of
// 8x-redundant global loads.
// ---------------------------------------------------------------------------
__global__ __launch_bounds__(256)
void gate_kernel(const unsigned short* __restrict__ xp,
                 const unsigned short* __restrict__ hp,
                 const unsigned short* __restrict__ Wg,
                 const float* __restrict__ bias,
                 float* __restrict__ cbuf,
                 unsigned short* __restrict__ hout) {
    __shared__ __align__(16) unsigned short tile[3 * 18 * 128]; // 13824 B
    __shared__ float zsh[16 * NG];                              // 16 KB

    const int r     = blockIdx.x / 3;
    const int cbase = (blockIdx.x % 3) * 16;
    const int tid   = threadIdx.x;
    const int lane  = tid & 31;
    const int w     = tid >> 5;
    const int m     = lane & 15;
    const int klo   = (lane & 16) >> 1;            // 0 or 8 (A-fragment half)

    // ---- stage combined [x|h] halo tile: rows r-1..r+1, cols cbase-1..cbase+16
    // tile layout: [rowo(3)][cidx(18)][ch128] bf16, ch<64 from x, ch>=64 from h.
    for (int i = tid; i < 864; i += 256) {          // 864 uint4 = 13824 B
        int elem = i * 8;
        int ch8  = elem & 127;
        int pc   = elem >> 7;
        int cidx = pc % 18;
        int rowo = pc / 18;
        const unsigned short* src = (ch8 < 64) ? xp : hp;
        v4u d = *(const v4u*)(src + ((r + rowo) * WP + (cbase + cidx)) * CC + (ch8 & 63));
        *(v4u*)(tile + elem) = d;
    }
    __syncthreads();

    v8f acc0 = {};
    v8f acc1 = {};

    const unsigned short* bbase = Wg + (size_t)(w * 2) * 512 + lane * 16;

    #pragma unroll 4
    for (int kt = 0; kt < 36; ++kt) {
        int tap   = kt >> 2;
        int dy    = tap / 3;
        int dx    = tap % 3;
        int ch128 = (kt & 3) * 32 + klo;

        const unsigned short* ap = tile + ((dy * 18 + (m + dx)) * 128 + ch128);
        ABFrag a;
        a.u[0] = *(const v4u*)ap;                  // K klo..klo+7
        a.u[1] = *(const v4u*)(ap + 16);           // K klo+16..klo+23

        const unsigned short* bp = bbase + (size_t)(kt * 16) * 512;
        ABFrag b0, b1;
        b0.u[0] = *(const v4u*)bp;
        b0.u[1] = *(const v4u*)(bp + 8);
        b1.u[0] = *(const v4u*)(bp + 512);
        b1.u[1] = *(const v4u*)(bp + 520);

        acc0 = __builtin_amdgcn_wmma_f32_16x16x32_bf16(
            false, a.v, false, b0.v, (short)0, acc0, false, false);
        acc1 = __builtin_amdgcn_wmma_f32_16x16x32_bf16(
            false, a.v, false, b1.v, (short)0, acc1, false, false);
    }

    // C/D layout: VGPR j -> M = j + (lane>=16 ? 8 : 0), N = lane&15 (+offset)
    const int mrow = (lane >> 4) * 8;
    const int n0   = w * 32 + (lane & 15);
    #pragma unroll
    for (int j = 0; j < 8; ++j) {
        zsh[(mrow + j) * NG + n0]      = acc0[j];
        zsh[(mrow + j) * NG + n0 + 16] = acc1[j];
    }
    __syncthreads();

    // LSTM pointwise: 16 px * 64 hidden channels = 1024 items / 256 threads
    #pragma unroll
    for (int it = 0; it < 4; ++it) {
        int idx = it * 256 + tid;
        int p   = idx >> 6;
        int hc  = idx & 63;
        float zi = zsh[p * NG + hc]        + bias[hc];
        float zf = zsh[p * NG + 64  + hc]  + bias[64 + hc];
        float zg = zsh[p * NG + 128 + hc]  + bias[128 + hc];
        float zo = zsh[p * NG + 192 + hc]  + bias[192 + hc];
        float gi = hsig(zi), gf = hsig(zf), go = hsig(zo);
        int col = cbase + p;
        int ci  = (r * WW + col) * HID + hc;
        float cn = gf * cbuf[ci] + gi * tanhf(zg);
        float hn = go * tanhf(cn);
        cbuf[ci] = cn;
        hout[((r + 1) * WP + (col + 1)) * CC + hc] = f2bf(hn);
    }
}

// ---------------------------------------------------------------------------
// Output kernel: y = conv(h_new, conv_w)+b, LayerNorm(ch), leaky ReLU.
// Block = 16-pixel segment, 4 waves each own 16 of the 64 output channels.
// h halo tile (3x18x64 bf16) staged in LDS, A fragments via ds_load_b128.
// ---------------------------------------------------------------------------
__global__ __launch_bounds__(128)
void out_kernel(const unsigned short* __restrict__ hp,
                const unsigned short* __restrict__ Wy,
                const float* __restrict__ convb,
                const float* __restrict__ gamma,
                const float* __restrict__ beta,
                float* __restrict__ out,
                unsigned short* __restrict__ xp,
                int step) {
    __shared__ __align__(16) unsigned short tile[3 * 18 * 64]; // 6912 B
    __shared__ float ysh[16 * NY];                             // 4 KB

    const int r     = blockIdx.x / 3;
    const int cbase = (blockIdx.x % 3) * 16;
    const int tid   = threadIdx.x;
    const int lane  = tid & 31;
    const int w     = tid >> 5;
    const int m     = lane & 15;
    const int klo   = (lane & 16) >> 1;

    // ---- stage h halo tile: [rowo(3)][cidx(18)][ch(64)]
    for (int i = tid; i < 432; i += 128) {          // 432 uint4 = 6912 B
        int elem = i * 8;
        int ch8  = elem & 63;
        int pc   = elem >> 6;
        int cidx = pc % 18;
        int rowo = pc / 18;
        v4u d = *(const v4u*)(hp + ((r + rowo) * WP + (cbase + cidx)) * CC + ch8);
        *(v4u*)(tile + elem) = d;
    }
    __syncthreads();

    v8f acc = {};

    #pragma unroll 2
    for (int kt = 0; kt < 18; ++kt) {
        int tap = kt >> 1;
        int dy  = tap / 3;
        int dx  = tap % 3;
        int ch  = (kt & 1) * 32 + klo;

        const unsigned short* ap = tile + ((dy * 18 + (m + dx)) * 64 + ch);
        ABFrag a;
        a.u[0] = *(const v4u*)ap;
        a.u[1] = *(const v4u*)(ap + 16);

        const unsigned short* bp = Wy + (size_t)(kt * 4 + w) * 512 + lane * 16;
        ABFrag b;
        b.u[0] = *(const v4u*)bp;
        b.u[1] = *(const v4u*)(bp + 8);

        acc = __builtin_amdgcn_wmma_f32_16x16x32_bf16(
            false, a.v, false, b.v, (short)0, acc, false, false);
    }

    const int mrow = (lane >> 4) * 8;
    const int n    = w * 16 + (lane & 15);
    #pragma unroll
    for (int j = 0; j < 8; ++j)
        ysh[(mrow + j) * NY + n] = acc[j] + convb[n];
    __syncthreads();

    if (tid < 16) {
        int p = tid;
        float s = 0.f, s2 = 0.f;
        #pragma unroll 8
        for (int c = 0; c < NY; ++c) {
            float v = ysh[p * NY + c];
            s += v; s2 += v * v;
        }
        float mu  = s * (1.f / NY);
        float var = s2 * (1.f / NY) - mu * mu;
        float inv = rsqrtf(var + 1e-3f);
        int col = cbase + p;
        int ob  = ((step * (HH * WW)) + (r * WW + col)) * NY;
        int xi  = ((r + 1) * WP + (col + 1)) * CC;
        #pragma unroll 8
        for (int c = 0; c < NY; ++c) {
            float v = gamma[c] * (ysh[p * NY + c] - mu) * inv + beta[c];
            v = (v >= 0.f) ? v : 0.2f * v;
            out[ob + c] = v;
            xp[xi + c]  = f2bf(v);
        }
    }
}

// ---------------------------------------------------------------------------
extern "C" void kernel_launch(void* const* d_in, const int* in_sizes, int n_in,
                              void* d_out, int out_size, void* d_ws, size_t ws_size,
                              hipStream_t stream) {
    const float* x0    = (const float*)d_in[0];
    const float* h0    = (const float*)d_in[1];
    const float* c0    = (const float*)d_in[2];
    const float* Wx    = (const float*)d_in[3];
    const float* Wh    = (const float*)d_in[4];
    const float* bias  = (const float*)d_in[5];
    const float* convw = (const float*)d_in[6];
    const float* convb = (const float*)d_in[7];
    const float* gamma = (const float*)d_in[8];
    const float* beta  = (const float*)d_in[9];
    float* out = (float*)d_out;

    char* ws = (char*)d_ws;
    unsigned short* Wg    = (unsigned short*)(ws);             // 1152*256*2 = 589824
    unsigned short* Wy    = (unsigned short*)(ws + 589824);    // 576*64*2   =  73728
    unsigned short* xpad  = (unsigned short*)(ws + 663552);    // 50*50*64*2 = 320000
    unsigned short* hpad0 = (unsigned short*)(ws + 983552);    // 320000
    unsigned short* hpad1 = (unsigned short*)(ws + 1303552);   // 320000
    float*          cbuf  = (float*)(ws + 1623552);            // 48*48*64*4 = 589824

    pack_gate_w<<<(KG * NG + 255) / 256, 256, 0, stream>>>(Wx, Wh, Wg);
    pack_out_w<<<(KY * NY + 255) / 256, 256, 0, stream>>>(convw, Wy);
    init_state<<<(HP * WP * CC + 255) / 256, 256, 0, stream>>>(
        x0, h0, c0, xpad, hpad0, hpad1, cbuf);

    const int blocks = HH * 3;   // 48 rows * 3 sixteen-pixel segments
    for (int t = 0; t < HORIZON; ++t) {
        const unsigned short* hin = (t & 1) ? hpad1 : hpad0;
        unsigned short*      hnew = (t & 1) ? hpad0 : hpad1;
        gate_kernel<<<blocks, 256, 0, stream>>>(xpad, hin, Wg, bias, cbuf, hnew);
        out_kernel<<<blocks, 128, 0, stream>>>(hnew, Wy, convb, gamma, beta,
                                               out, xpad, t);
    }
}
